// DynamicGroupAttention_77833397338377
// MI455X (gfx1250) — matrix operand order repaired and checked by reading, not compiled, not run on hardware
//
#include <hip/hip_runtime.h>
#include <cstdint>
#include <cstddef>

// ---------------------------------------------------------------------------
// DynamicGroupAttention for MI455X (gfx1250), wave32 + WMMA bf16 + TDM path.
// B=64, N=577 (Nq=576), D=768, H=12, DH=64, C=16, K=5, M=C*K=80
// ---------------------------------------------------------------------------

typedef __bf16 v16bf __attribute__((ext_vector_type(16)));
typedef float  v8f   __attribute__((ext_vector_type(8)));
typedef unsigned short u16x8 __attribute__((ext_vector_type(8)));
typedef unsigned int v4u __attribute__((ext_vector_type(4)));
typedef int v8i __attribute__((ext_vector_type(8)));
typedef int v4i __attribute__((ext_vector_type(4)));

union Frag {
    v16bf bf;
    u16x8 h[2];
    unsigned short u[16];
};

__device__ __forceinline__ unsigned short f32_to_bf16(float f) {
    union { float f; uint32_t u; } v; v.f = f;
    uint32_t b = v.u;
    uint32_t r = (b + 0x7FFFu + ((b >> 16) & 1u)) >> 16; // round-nearest-even
    return (unsigned short)r;
}

// Tensor Data Mover availability (this toolchain: 6-arg builtin, clang-23 lane)
#if defined(__has_builtin)
#if __has_builtin(__builtin_amdgcn_tensor_load_to_lds) && \
    __has_builtin(__builtin_amdgcn_s_wait_tensorcnt)
#define USE_TDM 1
#endif
#endif
#ifndef USE_TDM
#define USE_TDM 0
#endif

#if USE_TDM
// 2D TDM tile load: rows x rowlen bf16 elements, global row stride `stride`
// elements, packed contiguously into LDS at lds byte offset.
// D# packing per CDNA5 ISA.
__device__ __forceinline__ void tdm_load_tile(void* lds_ptr,
                                              const unsigned short* gptr,
                                              int rows, int rowlen, int stride) {
    unsigned long long ga = (unsigned long long)(uintptr_t)gptr;
    unsigned lds_off = (unsigned)(uintptr_t)lds_ptr; // generic LDS addr[31:0] = LDS offset
    v4u g0;
    g0[0] = 1u;                                           // count=1, user desc
    g0[1] = lds_off;                                      // lds_addr (bytes)
    g0[2] = (unsigned)(ga & 0xFFFFFFFFu);                 // global_addr[31:0]
    g0[3] = (unsigned)((ga >> 32) & 0x1FFFFFFu) | (2u << 30); // addr[56:32] | type=2
    unsigned td0 = (unsigned)stride;                      // tensor_dim0 (elems)
    unsigned td1 = 0x100000u;                             // tensor_dim1: large, no OOB
    v8i g1;
    g1[0] = (int)(1u << 16);                              // wg_mask=0 | data_size=1 (2B)
    g1[1] = (int)((td0 & 0xFFFFu) << 16);                 // abar=0 | tensor_dim0[15:0]
    g1[2] = (int)(((td0 >> 16) & 0xFFFFu) | ((td1 & 0xFFFFu) << 16));
    g1[3] = (int)(((td1 >> 16) & 0xFFFFu) | (((unsigned)rowlen & 0xFFFFu) << 16)); // tile_dim0
    g1[4] = (int)((unsigned)rows & 0xFFFFu);              // tile_dim1 | tile_dim2=0
    g1[5] = (int)((unsigned)stride);                      // tensor_dim0_stride[31:0]
    g1[6] = 0;                                            // stride[47:32] | dim1_stride lo
    g1[7] = 0;
    v4i g2 = {0, 0, 0, 0};
    v4i g3 = {0, 0, 0, 0};
    v8i g4 = {0, 0, 0, 0, 0, 0, 0, 0};                    // unused 5th group (zero)
    __builtin_amdgcn_tensor_load_to_lds(g0, g1, g2, g3, g4, 0);
}
#endif

// ---------------------------------------------------------------------------
// 1) x (f32, B x 577 x 768) -> xq bf16 (B x 576 x 768); CLS row -> d_out
// ---------------------------------------------------------------------------
__global__ void convert_x_kernel(const float* __restrict__ x,
                                 unsigned short* __restrict__ xq,
                                 float* __restrict__ out) {
    size_t i = (size_t)blockIdx.x * blockDim.x + threadIdx.x;
    const size_t total = (size_t)64 * 577 * 768;
    if (i >= total) return;
    int d = (int)(i % 768);
    size_t t = i / 768;
    int n = (int)(t % 577);
    int b = (int)(t / 577);
    float v = x[i];
    if (n == 0) {
        out[i] = v;                                   // passthrough CLS token
    } else {
        xq[((size_t)b * 576 + (n - 1)) * 768 + d] = f32_to_bf16(v);
    }
}

// W (f32, K x N row-major) -> Wt bf16 (N x K row-major)
__global__ void transpose_w_kernel(const float* __restrict__ W,
                                   unsigned short* __restrict__ Wt) {
    size_t i = (size_t)blockIdx.x * blockDim.x + threadIdx.x;
    if (i >= (size_t)768 * 768) return;
    int k = (int)(i % 768);
    int n = (int)(i / 768);
    Wt[i] = f32_to_bf16(W[(size_t)k * 768 + n]);
}

// ---------------------------------------------------------------------------
// 2) k-means: one 576-thread workgroup per batch (1 thread / token).
//    Dynamic LDS: cent[16*768] | sums[16*768] | cnt[16] | cnorm[16] (~98 KB,
//    legal: 320 KB LDS per WGP on CDNA5).
// ---------------------------------------------------------------------------
__global__ __launch_bounds__(576)
void kmeans_kernel(const float* __restrict__ x,
                   float* __restrict__ d2all,   // B x 576 x 16
                   int* __restrict__ assignv) { // B x 576
    extern __shared__ float smem[];
    float* cent  = smem;
    float* sums  = cent + 16 * 768;
    float* cnt   = sums + 16 * 768;
    float* cnorm = cnt + 16;

    const int b   = blockIdx.x;
    const int tid = threadIdx.x;
    const float* xb = x + ((size_t)b * 577 + 1) * 768;   // xq rows
    const float* xr = xb + (size_t)tid * 768;            // this thread's token

    for (int i = tid; i < 16 * 768; i += 576) cent[i] = xb[i]; // init: first 16 pts
    __syncthreads();

    for (int it = 0; it < 5; ++it) {
        for (int i = tid; i < 16 * 768; i += 576) sums[i] = 0.0f;
        if (tid < 16) {
            float s = 0.0f;
            for (int d = 0; d < 768; ++d) { float v = cent[tid * 768 + d]; s += v * v; }
            cnorm[tid] = s;
            cnt[tid] = 0.0f;
        }
        __syncthreads();

        {
            float dot[16];
            #pragma unroll
            for (int c = 0; c < 16; ++c) dot[c] = 0.0f;
            for (int d = 0; d < 768; ++d) {
                float xv = xr[d];
                #pragma unroll
                for (int c = 0; c < 16; ++c) dot[c] += xv * cent[c * 768 + d];
            }
            int best = 0;
            float bd = cnorm[0] - 2.0f * dot[0];
            #pragma unroll
            for (int c = 1; c < 16; ++c) {
                float v = cnorm[c] - 2.0f * dot[c];
                if (v < bd) { bd = v; best = c; }   // first-min tie-break (argmin)
            }
            atomicAdd(&cnt[best], 1.0f);
            float* srow = sums + best * 768;
            for (int d = 0; d < 768; ++d) atomicAdd(&srow[d], xr[d]);
        }
        __syncthreads();

        for (int i = tid; i < 16 * 768; i += 576) {
            int c = i / 768;
            float denom = fmaxf(cnt[c], 1.0f);
            cent[i] = 0.5f * cent[i] + 0.5f * (sums[i] / denom); // LR = 0.5
        }
        __syncthreads();
    }

    // final pass: full d2 matrix + assignment with converged centroids
    if (tid < 16) {
        float s = 0.0f;
        for (int d = 0; d < 768; ++d) { float v = cent[tid * 768 + d]; s += v * v; }
        cnorm[tid] = s;
    }
    __syncthreads();

    {
        float xn = 0.0f;
        float dot[16];
        #pragma unroll
        for (int c = 0; c < 16; ++c) dot[c] = 0.0f;
        for (int d = 0; d < 768; ++d) {
            float xv = xr[d];
            xn += xv * xv;
            #pragma unroll
            for (int c = 0; c < 16; ++c) dot[c] += xv * cent[c * 768 + d];
        }
        float* drow = d2all + ((size_t)b * 576 + tid) * 16;
        int best = 0;
        float bd = 3.4e38f;
        #pragma unroll
        for (int c = 0; c < 16; ++c) {
            float v = xn - 2.0f * dot[c] + cnorm[c];
            drow[c] = v;
            if (v < bd) { bd = v; best = c; }
        }
        assignv[b * 576 + tid] = best;
    }
}

// ---------------------------------------------------------------------------
// 3) top-K=5 smallest masked d2 per (b, c); key = d2 + BIG if not assigned.
// ---------------------------------------------------------------------------
__global__ void topk_kernel(const float* __restrict__ d2all,
                            const int* __restrict__ assignv,
                            int* __restrict__ idx) {
    int t = blockIdx.x * blockDim.x + threadIdx.x;
    if (t >= 64 * 16) return;
    int b = t / 16, c = t % 16;
    float bd[5];
    int bi[5];
    #pragma unroll
    for (int j = 0; j < 5; ++j) { bd[j] = 3.4e38f; bi[j] = 0; }
    for (int n = 0; n < 576; ++n) {
        float key = d2all[((size_t)b * 576 + n) * 16 + c];
        if (assignv[b * 576 + n] != c) key += 1000000000.0f;
        if (key < bd[4]) {
            int j = 4;
            while (j > 0 && key < bd[j - 1]) { bd[j] = bd[j - 1]; bi[j] = bi[j - 1]; --j; }
            bd[j] = key; bi[j] = n;      // strict <: stable like lax.top_k
        }
    }
    #pragma unroll
    for (int j = 0; j < 5; ++j) idx[(b * 16 + c) * 5 + j] = bi[j];
}

// gather kv bf16 rows (B x 80 x 768)
__global__ void gather_kernel(const unsigned short* __restrict__ xq,
                              const int* __restrict__ idx,
                              unsigned short* __restrict__ kv) {
    size_t i = (size_t)blockIdx.x * blockDim.x + threadIdx.x;
    const size_t total = (size_t)64 * 80 * 768;
    if (i >= total) return;
    int d = (int)(i % 768);
    size_t t = i / 768;
    int m = (int)(t % 80);
    int b = (int)(t / 80);
    int n = idx[b * 80 + m];
    kv[i] = xq[((size_t)b * 576 + n) * 768 + d];
}

// ---------------------------------------------------------------------------
// 4) bf16 WMMA GEMM: C[M x 768] = A[M x K] * Bt[768 x K]^T
//    Wave tile 16(M) x 64(N); the block-shared 64x32 B tile is staged in LDS,
//    double-buffered by the Tensor Data Mover (tensor_load_to_lds +
//    s_wait_tensorcnt) when available, else a cooperative copy.
//    mode 0: bf16 out. mode 1: f32 + bias, rows remapped into d_out[:,1:].
// ---------------------------------------------------------------------------
__global__ __launch_bounds__(256)
void gemm_bf16(const unsigned short* __restrict__ A,
               const unsigned short* __restrict__ Bt,
               unsigned short* __restrict__ Cb,
               float* __restrict__ Cf,
               const float* __restrict__ bias,
               int Mrows, int Kdim, int mode) {
    __shared__ unsigned short Btile[2][64 * 32];

    const int lane = threadIdx.x & 31;
    const int wave = threadIdx.x >> 5;
    const int half = lane >> 4;          // 0: lanes 0-15, 1: lanes 16-31
    const int l16  = lane & 15;
    const int n0 = blockIdx.x * 64;
    const int m0 = (blockIdx.y * 8 + wave) * 16;

    v8f z = {};
    v8f acc[4];
    #pragma unroll
    for (int a = 0; a < 4; ++a) acc[a] = z;

    const unsigned short* Arow  = A + (size_t)(m0 + l16) * Kdim;
    const unsigned short* Bbase = Bt + (size_t)n0 * Kdim;

#if USE_TDM
    if (wave == 0) {                     // 2-deep TDM pipeline
        tdm_load_tile(&Btile[0][0], Bbase, 64, 32, Kdim);
        tdm_load_tile(&Btile[1][0], Bbase + 32, 64, 32, Kdim);
    }
#endif

    for (int k0 = 0; k0 < Kdim; k0 += 32) {
        const int cur = (k0 >> 5) & 1;
#if USE_TDM
        if (wave == 0) {                 // wait for oldest in-flight TDM (buf[cur])
            if (k0 + 32 < Kdim) __builtin_amdgcn_s_wait_tensorcnt(1);
            else                __builtin_amdgcn_s_wait_tensorcnt(0);
        }
        __syncthreads();                 // publish Btile[cur]
#else
        __syncthreads();
        {   // cooperative stage: 2048 elems = 256 x u16x8 chunks
            int row = threadIdx.x >> 2, seg = threadIdx.x & 3;
            *(u16x8*)&Btile[cur][row * 32 + seg * 8] =
                *(const u16x8*)(Bbase + (size_t)row * Kdim + k0 + seg * 8);
        }
        __syncthreads();
#endif
        if (k0 + 64 < Kdim) __builtin_prefetch(Arow + k0 + 64, 0, 1);

        Frag af;                                              // A 16x32 bf16: K contig
        af.h[0] = *(const u16x8*)(Arow + k0 + half * 16);     // K = k0+{0|16}+e
        af.h[1] = *(const u16x8*)(Arow + k0 + half * 16 + 8); // K = k0+{0|16}+8+e
        #pragma unroll
        for (int a = 0; a < 4; ++a) {
            const unsigned short* brow = &Btile[cur][(a * 16 + l16) * 32 + half * 16];
            Frag bfm;                                         // B 32x16: N=l16, K contig
            bfm.h[0] = *(const u16x8*)(brow);
            bfm.h[1] = *(const u16x8*)(brow + 8);
            acc[a] = __builtin_amdgcn_wmma_f32_16x16x32_bf16(
                false, af.bf, false, bfm.bf, (short)0, acc[a], false, false);
        }
#if USE_TDM
        if (k0 + 32 < Kdim) {
            __syncthreads();             // all waves done reading Btile[cur]
            if (wave == 0 && k0 + 64 < Kdim)
                tdm_load_tile(&Btile[cur][0], Bbase + k0 + 64, 64, 32, Kdim);
        }
#endif
    }

    if (mode == 0) {
        #pragma unroll
        for (int a = 0; a < 4; ++a)
            #pragma unroll
            for (int r = 0; r < 8; ++r) {
                int row = m0 + r + half * 8;
                int col = n0 + a * 16 + l16;
                Cb[(size_t)row * 768 + col] = f32_to_bf16(acc[a][r]);
            }
    } else {
        #pragma unroll
        for (int a = 0; a < 4; ++a)
            #pragma unroll
            for (int r = 0; r < 8; ++r) {
                int row = m0 + r + half * 8;
                int col = n0 + a * 16 + l16;
                int bb = row / 576, nn = row % 576;
                Cf[((size_t)bb * 577 + 1 + nn) * 768 + col] = acc[a][r] + bias[col];
            }
    }
    (void)Mrows;
}

// ---------------------------------------------------------------------------
// 5) Attention: one block per (b, h). K and V^T panels staged in LDS (shared
//    by all 8 waves x 36 M-tiles); scores(576x80) via WMMA; in-register
//    softmax (16-lane shfl matches C/D layout); attn@V via WMMA, K padded 96.
// ---------------------------------------------------------------------------
__global__ __launch_bounds__(256)
void attention_kernel(const unsigned short* __restrict__ Q,   // 36864 x 768
                      const unsigned short* __restrict__ Kb,  // 5120 x 768
                      const unsigned short* __restrict__ Vb,  // 5120 x 768
                      unsigned short* __restrict__ AO) {      // 36864 x 768
    __shared__ unsigned short kT[80 * 64];          // [m][d] row-major
    __shared__ unsigned short vT[64 * 96];          // [d][m], m padded 80->96
    __shared__ unsigned short attnS[8 * 16 * 96];   // per-wave 16x96 bf16

    const int bh = blockIdx.x;
    const int b = bh / 12, h = bh % 12;
    const int lane = threadIdx.x & 31;
    const int wave = threadIdx.x >> 5;
    const int half = lane >> 4;
    const int l16  = lane & 15;

    // stage K panel (vectorized) and V^T panel
    for (int ch = threadIdx.x; ch < 80 * 8; ch += 256) {
        int row = ch >> 3, seg = ch & 7;
        *(u16x8*)&kT[row * 64 + seg * 8] =
            *(const u16x8*)(Kb + ((size_t)b * 80 + row) * 768 + h * 64 + seg * 8);
    }
    for (int i = threadIdx.x; i < 64 * 96; i += 256) {
        int d = i / 96, m = i % 96;
        unsigned short val = 0;
        if (m < 80) val = Vb[((size_t)b * 80 + m) * 768 + h * 64 + d];
        vT[d * 96 + m] = val;
    }
    __syncthreads();

    unsigned short* aw = attnS + wave * 16 * 96;
    v8f z = {};

    for (int mt = wave; mt < 36; mt += 8) {
        const int m0 = mt * 16;
        const unsigned short* qrow = Q + ((size_t)b * 576 + m0 + l16) * 768 + h * 64;

        // ---- scores = q . k^T  (16 x 80, K = 64) ----
        v8f sc[5];
        #pragma unroll
        for (int a = 0; a < 5; ++a) sc[a] = z;
        #pragma unroll
        for (int k0 = 0; k0 < 64; k0 += 32) {
            Frag af;
            af.h[0] = *(const u16x8*)(qrow + k0 + half * 16);
            af.h[1] = *(const u16x8*)(qrow + k0 + half * 16 + 8);
            #pragma unroll
            for (int a = 0; a < 5; ++a) {
                const unsigned short* krow = &kT[(a * 16 + l16) * 64 + k0 + half * 16];
                Frag bfm;
                bfm.h[0] = *(const u16x8*)(krow);
                bfm.h[1] = *(const u16x8*)(krow + 8);
                sc[a] = __builtin_amdgcn_wmma_f32_16x16x32_bf16(
                    false, af.bf, false, bfm.bf, (short)0, sc[a], false, false);
            }
        }

        // ---- softmax over 80 per row (row = r + half*8, cols across l16) ----
        const float scale = 0.125f;                       // DH^-0.5
        #pragma unroll
        for (int r = 0; r < 8; ++r) {
            float mx = -3.4e38f;
            #pragma unroll
            for (int a = 0; a < 5; ++a) {
                float v = sc[a][r] * scale;
                sc[a][r] = v;
                mx = fmaxf(mx, v);
            }
            for (int off = 1; off < 16; off <<= 1)
                mx = fmaxf(mx, __shfl_xor(mx, off, 32));  // stays in 16-lane half
            float sum = 0.0f;
            #pragma unroll
            for (int a = 0; a < 5; ++a) {
                float e = __expf(sc[a][r] - mx);
                sc[a][r] = e;
                sum += e;
            }
            for (int off = 1; off < 16; off <<= 1)
                sum += __shfl_xor(sum, off, 32);
            float inv = 1.0f / sum;
            int row = r + half * 8;
            #pragma unroll
            for (int a = 0; a < 5; ++a)
                aw[row * 96 + a * 16 + l16] = f32_to_bf16(sc[a][r] * inv);
            aw[row * 96 + 80 + l16] = 0;                  // zero pad K=80..95
        }
        // per-wave LDS region; same-wave DS ops stay in order

        // ---- out = attn(16x96) @ Vpad(96x64) ----
        v8f outv[4];
        #pragma unroll
        for (int a = 0; a < 4; ++a) outv[a] = z;
        const unsigned short* arow = aw + l16 * 96;       // A row M = l16
        #pragma unroll
        for (int k0 = 0; k0 < 96; k0 += 32) {
            Frag af;
            af.h[0] = *(const u16x8*)(arow + k0 + half * 16);
            af.h[1] = *(const u16x8*)(arow + k0 + half * 16 + 8);
            #pragma unroll
            for (int a = 0; a < 4; ++a) {
                const unsigned short* vrow = &vT[(a * 16 + l16) * 96 + k0 + half * 16];
                Frag bfm;
                bfm.h[0] = *(const u16x8*)(vrow);
                bfm.h[1] = *(const u16x8*)(vrow + 8);
                outv[a] = __builtin_amdgcn_wmma_f32_16x16x32_bf16(
                    false, af.bf, false, bfm.bf, (short)0, outv[a], false, false);
            }
        }

        #pragma unroll
        for (int a = 0; a < 4; ++a)
            #pragma unroll
            for (int r = 0; r < 8; ++r) {
                int row = m0 + r + half * 8;
                AO[((size_t)b * 576 + row) * 768 + h * 64 + a * 16 + l16] =
                    f32_to_bf16(outv[a][r]);
            }
    }
}

// ---------------------------------------------------------------------------
// Host orchestration
// ---------------------------------------------------------------------------
extern "C" void kernel_launch(void* const* d_in, const int* in_sizes, int n_in,
                              void* d_out, int out_size, void* d_ws, size_t ws_size,
                              hipStream_t stream) {
    (void)in_sizes; (void)n_in; (void)out_size; (void)ws_size;
    const float* x  = (const float*)d_in[0];
    const float* Wq = (const float*)d_in[1];
    const float* Wk = (const float*)d_in[2];
    const float* Wv = (const float*)d_in[3];
    const float* Wo = (const float*)d_in[4];
    const float* bo = (const float*)d_in[5];
    float* out = (float*)d_out;

    char* ws = (char*)d_ws;
    size_t off = 0;
    auto alloc = [&](size_t bytes) -> char* {
        char* p = ws + off;
        off = (off + bytes + 255) & ~(size_t)255;
        return p;
    };

    unsigned short* xqb = (unsigned short*)alloc((size_t)64 * 576 * 768 * 2);
    unsigned short* wqT = (unsigned short*)alloc((size_t)768 * 768 * 2);
    unsigned short* wkT = (unsigned short*)alloc((size_t)768 * 768 * 2);
    unsigned short* wvT = (unsigned short*)alloc((size_t)768 * 768 * 2);
    unsigned short* woT = (unsigned short*)alloc((size_t)768 * 768 * 2);
    float* d2all        = (float*)alloc((size_t)64 * 576 * 16 * 4);
    int* assignv        = (int*)alloc((size_t)64 * 576 * 4);
    int* idx            = (int*)alloc((size_t)64 * 80 * 4);
    unsigned short* kvb = (unsigned short*)alloc((size_t)64 * 80 * 768 * 2);
    unsigned short* qb  = (unsigned short*)alloc((size_t)64 * 576 * 768 * 2);
    unsigned short* kb  = (unsigned short*)alloc((size_t)64 * 80 * 768 * 2);
    unsigned short* vb  = (unsigned short*)alloc((size_t)64 * 80 * 768 * 2);
    unsigned short* aob = (unsigned short*)alloc((size_t)64 * 576 * 768 * 2);

    {
        size_t total = (size_t)64 * 577 * 768;
        convert_x_kernel<<<(unsigned)((total + 255) / 256), 256, 0, stream>>>(x, xqb, out);
    }
    transpose_w_kernel<<<2304, 256, 0, stream>>>(Wq, wqT);
    transpose_w_kernel<<<2304, 256, 0, stream>>>(Wk, wkT);
    transpose_w_kernel<<<2304, 256, 0, stream>>>(Wv, wvT);
    transpose_w_kernel<<<2304, 256, 0, stream>>>(Wo, woT);

    kmeans_kernel<<<64, 576, (16 * 768 * 2 + 32) * sizeof(float), stream>>>(x, d2all, assignv);
    topk_kernel<<<4, 256, 0, stream>>>(d2all, assignv, idx);
    {
        size_t total = (size_t)64 * 80 * 768;
        gather_kernel<<<(unsigned)((total + 255) / 256), 256, 0, stream>>>(xqb, idx, kvb);
    }

    gemm_bf16<<<dim3(12, 288), 256, 0, stream>>>(xqb, wqT, qb, nullptr, nullptr, 36864, 768, 0);
    gemm_bf16<<<dim3(12, 40),  256, 0, stream>>>(kvb, wkT, kb, nullptr, nullptr, 5120, 768, 0);
    gemm_bf16<<<dim3(12, 40),  256, 0, stream>>>(kvb, wvT, vb, nullptr, nullptr, 5120, 768, 0);

    attention_kernel<<<768, 256, 0, stream>>>(qb, kb, vb, aob);

    gemm_bf16<<<dim3(12, 288), 256, 0, stream>>>(aob, woT, nullptr, out, bo, 36864, 768, 1);
}
